// EarlyFusionModel_19980187861743
// MI455X (gfx1250) — matrix operand order repaired and checked by reading, not compiled, not run on hardware
//
#include <hip/hip_runtime.h>

// ---------------------------------------------------------------------------
// CDNA5 (gfx1250) EarlyFusionModel: GATv2 x2 -> maxpool -> fused MLP head.
// GEMMs: v_wmma_f32_16x16x32_f16 with pre-transposed, zero-padded f16 weights
// (B fragment = one contiguous 32B load, no bounds checks in the K loop).
// Weight prep kernel stages W tiles into LDS via the Tensor Data Mover
// (tensor_load_to_lds + s_wait_tensorcnt) when the builtin is available.
// Edge softmax/aggregation path uses hw float atomics (global_atomic_add_f32).
// ---------------------------------------------------------------------------

typedef __attribute__((ext_vector_type(16))) _Float16 v16h;
typedef __attribute__((ext_vector_type(8)))  float    v8f;
typedef __attribute__((ext_vector_type(4)))  unsigned v4u;
typedef __attribute__((ext_vector_type(8)))  int      v8i;
typedef __attribute__((ext_vector_type(4)))  int      v4i;

namespace {
constexpr int kN    = 32768;
constexpr int kE    = 262144;
constexpr int kG    = 1024;
constexpr int kFin  = 32;
constexpr int kHC   = 256;     // 4 heads x 64
constexpr int kEtot = kE + kN; // edges + self loops
constexpr float kBnInv = 0.99999500003749977f; // 1/sqrt(1+1e-5)
}

#if defined(__HIP_DEVICE_COMPILE__) && __has_builtin(__builtin_amdgcn_tensor_load_to_lds)
#define USE_TDM 1
#endif

// ---------------- small helpers -------------------------------------------

__device__ __forceinline__ void atomicAddF(float* p, float v) {
  unsafeAtomicAdd(p, v);                     // -> global_atomic_add_f32
}

// monotone float<->uint key for atomicMax-based segment max (handles sign)
__device__ __forceinline__ unsigned fkey(float f) {
  unsigned u = __float_as_uint(f);
  return u ^ (unsigned)(((int)u >> 31) | 0x80000000);
}
__device__ __forceinline__ float funkey(unsigned u) {
  unsigned m = ((u >> 31) - 1u) | 0x80000000u;
  return __uint_as_float(u ^ m);
}

__device__ __forceinline__ void ld8(const float* __restrict__ p, float* v) {
  float4 a = ((const float4*)p)[0];
  float4 b = ((const float4*)p)[1];
  v[0]=a.x; v[1]=a.y; v[2]=a.z; v[3]=a.w;
  v[4]=b.x; v[5]=b.y; v[6]=b.z; v[7]=b.w;
}

// ---------------- fills ----------------------------------------------------

__global__ __launch_bounds__(256) void k_fillf(float* p, float v, long long n) {
  long long i = (long long)blockIdx.x * blockDim.x + threadIdx.x;
  long long s = (long long)gridDim.x * blockDim.x;
  for (; i < n; i += s) p[i] = v;
}
__global__ __launch_bounds__(256) void k_fillu(unsigned* p, unsigned v, long long n) {
  long long i = (long long)blockIdx.x * blockDim.x + threadIdx.x;
  long long s = (long long)gridDim.x * blockDim.x;
  for (; i < n; i += s) p[i] = v;
}

// ---------------- weight prep: W[KxN] f32 -> Wt[Np x Kp] f16 (transposed) --
// One block handles 32 K-rows. TDM stages the f32 tile into LDS, then threads
// convert + transpose to global. Wt arena must be pre-zeroed (padding).

__global__ __launch_bounds__(256) void k_wprep(
    const float* __restrict__ W, _Float16* __restrict__ Wt,
    int K, int Nout, int Kp)
{
  __shared__ float smem[32 * 256];          // 32 rows x up to 256 cols
  const int r0 = blockIdx.x * 32;
#if defined(USE_TDM)
  if (threadIdx.x < 32) {                   // one wave issues the TDM op
    unsigned long long ga =
        (unsigned long long)W + (unsigned long long)r0 * Nout * 4ull;
    unsigned lds = (unsigned)(unsigned long long)(&smem[0]); // low 32 = LDS off
    v4u g0;
    g0[0] = 1u;                                             // count=1
    g0[1] = lds;                                            // lds_addr
    g0[2] = (unsigned)(ga & 0xFFFFFFFFull);                 // global_addr lo
    g0[3] = (unsigned)((ga >> 32) & 0x1FFFFFFull) | (2u << 30); // hi + type=2
    unsigned d0 = (unsigned)Nout;          // tensor_dim0
    unsigned d1 = (unsigned)(K - r0);      // remaining rows (OOB reads -> 0)
    unsigned s0 = (unsigned)Nout;          // tensor_dim0_stride
    v8i g1;
    g1[0] = (int)(2u << 16);                                // data_size = 4B
    g1[1] = (int)((d0 & 0xFFFFu) << 16);                    // dim0[15:0]
    g1[2] = (int)((d0 >> 16) | ((d1 & 0xFFFFu) << 16));     // dim0hi|dim1lo
    g1[3] = (int)((d1 >> 16) | ((unsigned)Nout << 16));     // dim1hi|tile0
    g1[4] = (int)32u;                                       // tile_dim1 = 32
    g1[5] = (int)s0;                                        // stride0 lo32
    g1[6] = 0;                                              // stride0 hi16
    g1[7] = 0;
    v4i gz = {};
#if __clang_major__ >= 23
    v8i gz8 = {};
    __builtin_amdgcn_tensor_load_to_lds(g0, g1, gz, gz, gz8, 0);
#else
    __builtin_amdgcn_tensor_load_to_lds(g0, g1, gz, gz, 0);
#endif
    __builtin_amdgcn_s_wait_tensorcnt(0);
  }
  __syncthreads();
#else
  for (int i = threadIdx.x; i < 32 * Nout; i += 256) {
    int k = r0 + i / Nout, c = i % Nout;
    smem[i] = (k < K) ? W[(size_t)k * Nout + c] : 0.f;
  }
  __syncthreads();
#endif
  for (int i = threadIdx.x; i < 32 * Nout; i += 256) {
    int kk = i / Nout, c = i % Nout;
    int k = r0 + kk;
    if (k < K) Wt[(size_t)c * Kp + k] = (_Float16)smem[(size_t)kk * Nout + c];
  }
}

// ---------------- WMMA GEMM: C = epi(A[MxK] @ W + bias) --------------------
// Wt is [Np x Kp] f16, row c = column c of W, zero padded. One wave owns a
// 16x64 output strip (4 accumulators share one A fragment).
// mode: 0 = +bias ; 1 = +bias,relu ; 2 = +bias,relu,BN(gamma*inv, beta)

__device__ __forceinline__ v16h a_frag_fast(const float* __restrict__ Arow,
                                            int kk, int hi) {
  float g0[8], g1[8];
  ld8(Arow + kk + hi * 16, g0);
  ld8(Arow + kk + hi * 16 + 8, g1);
  v16h a;
#pragma unroll
  for (int e = 0; e < 8; ++e) {
    a[e]     = (_Float16)g0[e];
    a[8 + e] = (_Float16)g1[e];
  }
  return a;
}

__global__ __launch_bounds__(256) void k_wmma_gemm(
    const float* __restrict__ A, const _Float16* __restrict__ Wt,
    const float* __restrict__ bias, const float* __restrict__ gamma,
    const float* __restrict__ beta, int M, int K, int Kp, int Nout,
    float* __restrict__ C, int ldc, int colOff, int mode)
{
  const int lane   = threadIdx.x & 31;
  const int wave   = threadIdx.x >> 5;
  const int tilesM = M >> 4;
  const int strips = (Nout + 63) >> 6;
  const int strip  = blockIdx.x * 8 + wave;
  if (strip >= tilesM * strips) return;
  const int tm  = strip % tilesM;
  const int tn0 = (strip / tilesM) * 4;
  const int lo  = lane & 15;
  const int hi  = lane >> 4;
  const float* __restrict__ Arow = A + (size_t)(tm * 16 + lo) * K;
  // lane's B stream: col = (tn0+t)*16+lo, elems K = kk + hi*16 + 0..15
  const _Float16* __restrict__ Bp = Wt + (size_t)(tn0 * 16 + lo) * Kp + hi * 16;
  const size_t ts = (size_t)16 * Kp;       // next 16-col tile

  v8f acc0 = {}, acc1 = {}, acc2 = {}, acc3 = {};
  const int K32 = K & ~31;
  int kk = 0;
  for (; kk < K32; kk += 32) {
    __builtin_prefetch(Arow + kk + 64, 0, 1);     // global_prefetch_b8
    v16h a  = a_frag_fast(Arow, kk, hi);
    v16h b0 = *(const v16h*)(Bp + kk);
    v16h b1 = *(const v16h*)(Bp + ts + kk);
    v16h b2 = *(const v16h*)(Bp + 2 * ts + kk);
    v16h b3 = *(const v16h*)(Bp + 3 * ts + kk);
    acc0 = __builtin_amdgcn_wmma_f32_16x16x32_f16(false, a, false, b0, (short)0, acc0, false, false);
    acc1 = __builtin_amdgcn_wmma_f32_16x16x32_f16(false, a, false, b1, (short)0, acc1, false, false);
    acc2 = __builtin_amdgcn_wmma_f32_16x16x32_f16(false, a, false, b2, (short)0, acc2, false, false);
    acc3 = __builtin_amdgcn_wmma_f32_16x16x32_f16(false, a, false, b3, (short)0, acc3, false, false);
  }
  if (kk < K) {                                   // K tail (only desc: K=200)
    v16h a;
#pragma unroll
    for (int e = 0; e < 8; ++e) {
      int k0 = kk + hi * 16 + e;
      int k1 = kk + hi * 16 + 8 + e;
      a[e]     = (k0 < K) ? (_Float16)Arow[k0] : (_Float16)0.f;
      a[8 + e] = (k1 < K) ? (_Float16)Arow[k1] : (_Float16)0.f;
    }
    v16h b0 = *(const v16h*)(Bp + kk);            // padded: in-range, zeros
    v16h b1 = *(const v16h*)(Bp + ts + kk);
    v16h b2 = *(const v16h*)(Bp + 2 * ts + kk);
    v16h b3 = *(const v16h*)(Bp + 3 * ts + kk);
    acc0 = __builtin_amdgcn_wmma_f32_16x16x32_f16(false, a, false, b0, (short)0, acc0, false, false);
    acc1 = __builtin_amdgcn_wmma_f32_16x16x32_f16(false, a, false, b1, (short)0, acc1, false, false);
    acc2 = __builtin_amdgcn_wmma_f32_16x16x32_f16(false, a, false, b2, (short)0, acc2, false, false);
    acc3 = __builtin_amdgcn_wmma_f32_16x16x32_f16(false, a, false, b3, (short)0, acc3, false, false);
  }
  // epilogue: VGPR r -> row tm*16 + r + 8*hi, col (tn0+t)*16 + lo
#pragma unroll
  for (int t = 0; t < 4; ++t) {
    v8f av = t == 0 ? acc0 : t == 1 ? acc1 : t == 2 ? acc2 : acc3;
    int col = (tn0 + t) * 16 + lo;
    if (col >= Nout) continue;
    float bb = bias ? bias[col] : 0.f;
    float g = 1.f, be = 0.f;
    if (mode == 2) { g = gamma[col] * kBnInv; be = beta[col]; }
#pragma unroll
    for (int r = 0; r < 8; ++r) {
      int orow = tm * 16 + r + 8 * hi;
      float v = av[r] + bb;
      if (mode >= 1) v = v > 0.f ? v : 0.f;
      if (mode == 2) v = v * g + be;
      C[(size_t)orow * ldc + colOff + col] = v;
    }
  }
}

// ---------------- GAT edge pipeline ----------------------------------------

__global__ __launch_bounds__(256) void k_deg_ea(
    const int* __restrict__ ei, const float* __restrict__ ea,
    float* __restrict__ deg, float* __restrict__ eas)
{
  int e = blockIdx.x * blockDim.x + threadIdx.x;
  if (e >= kE) return;
  int d = ei[kE + e];
  atomicAddF(&deg[d], 1.f);
#pragma unroll
  for (int k = 0; k < 8; ++k) atomicAddF(&eas[d * 8 + k], ea[(size_t)e * 8 + k]);
}

__global__ __launch_bounds__(256) void k_ea_div(
    const float* __restrict__ deg, float* __restrict__ eas)
{
  int i = blockIdx.x * blockDim.x + threadIdx.x;
  if (i >= kN * 8) return;
  eas[i] = eas[i] / fmaxf(deg[i >> 3], 1.f);
}

// wave per edge; lane handles 8 channels; head = lane>>3
__global__ __launch_bounds__(256) void k_edge_logits(
    const float* __restrict__ xl, const float* __restrict__ xr,
    const int* __restrict__ ei, const float* __restrict__ ea,
    const float* __restrict__ ealoop, const float* __restrict__ We,
    const float* __restrict__ att, float* __restrict__ logit)
{
  int gw = (blockIdx.x * blockDim.x + threadIdx.x) >> 5;
  if (gw >= kEtot) return;
  int lane = threadIdx.x & 31;
  int s, d; const float* eap;
  if (gw < kE) { s = ei[gw]; d = ei[kE + gw]; eap = ea + (size_t)gw * 8; }
  else         { s = d = gw - kE;             eap = ealoop + (size_t)(gw - kE) * 8; }
  float ev[8]; ld8(eap, ev);
  int c0 = lane * 8;
  float lv[8], rv[8];
  ld8(xl + (size_t)s * kHC + c0, lv);
  ld8(xr + (size_t)d * kHC + c0, rv);
  float partial = 0.f;
#pragma unroll
  for (int i = 0; i < 8; ++i) {
    int c = c0 + i;
    float e8 = 0.f;
#pragma unroll
    for (int k = 0; k < 8; ++k) e8 = fmaf(ev[k], We[k * kHC + c], e8);
    float m = lv[i] + rv[i] + e8;
    m = m > 0.f ? m : 0.2f * m;               // leaky_relu(0.2)
    partial = fmaf(m, att[c], partial);       // att flat [H*C] == channel idx
  }
  partial += __shfl_xor(partial, 1, 32);
  partial += __shfl_xor(partial, 2, 32);
  partial += __shfl_xor(partial, 4, 32);
  if ((lane & 7) == 0) logit[(size_t)gw * 4 + (lane >> 3)] = partial;
}

__global__ __launch_bounds__(256) void k_segmax(
    const float* __restrict__ logit, const int* __restrict__ ei,
    unsigned* __restrict__ lmax)
{
  int i = blockIdx.x * blockDim.x + threadIdx.x;
  if (i >= kEtot * 4) return;
  int e = i >> 2, h = i & 3;
  int d = (e < kE) ? ei[kE + e] : (e - kE);
  atomicMax(&lmax[d * 4 + h], fkey(logit[i]));
}

__global__ __launch_bounds__(256) void k_expsum(
    float* __restrict__ pbuf, const int* __restrict__ ei,
    const unsigned* __restrict__ lmax, float* __restrict__ psum)
{
  int i = blockIdx.x * blockDim.x + threadIdx.x;
  if (i >= kEtot * 4) return;
  int e = i >> 2, h = i & 3;
  int d = (e < kE) ? ei[kE + e] : (e - kE);
  float p = expf(pbuf[i] - funkey(lmax[d * 4 + h]));
  pbuf[i] = p;
  atomicAddF(&psum[d * 4 + h], p);
}

__global__ __launch_bounds__(256) void k_aggregate(
    const float* __restrict__ xl, const float* __restrict__ pbuf,
    const float* __restrict__ psum, const int* __restrict__ ei,
    float* __restrict__ agg)
{
  int gw = (blockIdx.x * blockDim.x + threadIdx.x) >> 5;
  if (gw >= kEtot) return;
  int lane = threadIdx.x & 31;
  int s, d;
  if (gw < kE) { s = ei[gw]; d = ei[kE + gw]; } else { s = d = gw - kE; }
  int head = lane >> 3;
  float alpha = pbuf[(size_t)gw * 4 + head] / psum[d * 4 + head];
  int c0 = lane * 8;
  float lv[8]; ld8(xl + (size_t)s * kHC + c0, lv);
#pragma unroll
  for (int i = 0; i < 8; ++i)
    atomicAddF(&agg[(size_t)d * kHC + c0 + i], lv[i] * alpha);
}

__global__ __launch_bounds__(256) void k_bias_relu(
    float* __restrict__ h, const float* __restrict__ bias, long long n)
{
  long long i = (long long)blockIdx.x * blockDim.x + threadIdx.x;
  if (i >= n) return;
  float v = h[i] + bias[i & (kHC - 1)];
  h[i] = v > 0.f ? v : 0.f;
}

// global max pool: h >= 0 (post-relu) so float bits compare as unsigned
__global__ __launch_bounds__(256) void k_pool(
    const float* __restrict__ h, const int* __restrict__ batch,
    float* __restrict__ xg)
{
  long long i = (long long)blockIdx.x * blockDim.x + threadIdx.x;
  if (i >= (long long)kN * kHC) return;
  int n = (int)(i >> 8), c = (int)(i & (kHC - 1));
  atomicMax((unsigned*)&xg[(size_t)batch[n] * kHC + c], __float_as_uint(h[i]));
}

// ---------------- host-side orchestration ---------------------------------

struct WT { const _Float16* p; int Kp; };

static void gemm(hipStream_t s, const float* A, WT wt, const float* b,
                 const float* gn, const float* bt, int M, int K, int Nout,
                 float* C, int ldc, int colOff, int mode)
{
  int tiles = (M >> 4) * ((Nout + 63) >> 6);
  k_wmma_gemm<<<(tiles + 7) / 8, 256, 0, s>>>(A, wt.p, b, gn, bt, M, K, wt.Kp,
                                              Nout, C, ldc, colOff, mode);
}

static void fillf(hipStream_t s, float* p, float v, long long n) {
  int blocks = (int)((n + 255) / 256);
  if (blocks > 1048576) blocks = 1048576;
  k_fillf<<<blocks, 256, 0, s>>>(p, v, n);
}
static void fillu(hipStream_t s, unsigned* p, unsigned v, long long n) {
  int blocks = (int)((n + 255) / 256);
  if (blocks > 1048576) blocks = 1048576;
  k_fillu<<<blocks, 256, 0, s>>>(p, v, n);
}

static void gat_layer(hipStream_t st, const float* in, int fin,
                      WT wl, WT wr, const float* const* w, // Wl,bl,Wr,br,We,att,bias
                      float* xl, float* xr, float* aggOut,
                      unsigned* lmax, float* psum, float* pbuf,
                      const int* ei, const float* ea, const float* ealoop)
{
  gemm(st, in, wl, w[1], nullptr, nullptr, kN, fin, kHC, xl, kHC, 0, 0);
  gemm(st, in, wr, w[3], nullptr, nullptr, kN, fin, kHC, xr, kHC, 0, 0);
  k_edge_logits<<<(kEtot * 32 + 255) / 256, 256, 0, st>>>(
      xl, xr, ei, ea, ealoop, w[4], w[5], pbuf);
  fillu(st, lmax, 0u, (long long)kN * 4);
  k_segmax<<<(kEtot * 4 + 255) / 256, 256, 0, st>>>(pbuf, ei, lmax);
  fillf(st, psum, 0.f, (long long)kN * 4);
  k_expsum<<<(kEtot * 4 + 255) / 256, 256, 0, st>>>(pbuf, ei, lmax, psum);
  fillf(st, aggOut, 0.f, (long long)kN * kHC);
  k_aggregate<<<(kEtot * 32 + 255) / 256, 256, 0, st>>>(xl, pbuf, psum, ei, aggOut);
  k_bias_relu<<<((long long)kN * kHC + 255) / 256, 256, 0, st>>>(
      aggOut, w[6], (long long)kN * kHC);
}

extern "C" void kernel_launch(void* const* d_in, const int* in_sizes, int n_in,
                              void* d_out, int out_size, void* d_ws, size_t ws_size,
                              hipStream_t stream)
{
  const float* x     = (const float*)d_in[0];
  const int*   ei    = (const int*)d_in[1];
  const float* ea    = (const float*)d_in[2];
  const int*   batch = (const int*)d_in[3];
  const float* fp    = (const float*)d_in[4];
  const float* desc  = (const float*)d_in[5];
  const float* c1w[7]; for (int i = 0; i < 7; ++i) c1w[i] = (const float*)d_in[6 + i];
  const float* c2w[7]; for (int i = 0; i < 7; ++i) c2w[i] = (const float*)d_in[13 + i];
  const float* gW  = (const float*)d_in[20], *gb  = (const float*)d_in[21];
  const float* ggn = (const float*)d_in[22], *gbt = (const float*)d_in[23];
  const float* fW  = (const float*)d_in[24], *fb  = (const float*)d_in[25];
  const float* fgn = (const float*)d_in[26], *fbt = (const float*)d_in[27];
  const float* dW  = (const float*)d_in[28], *db  = (const float*)d_in[29];
  const float* dgn = (const float*)d_in[30], *dbt = (const float*)d_in[31];
  const float* k1W = (const float*)d_in[32], *k1b = (const float*)d_in[33];
  const float* k1g = (const float*)d_in[34], *k1t = (const float*)d_in[35];
  const float* k2W = (const float*)d_in[36], *k2b = (const float*)d_in[37];
  const float* k2g = (const float*)d_in[38], *k2t = (const float*)d_in[39];
  const float* k3W = (const float*)d_in[40], *k3b = (const float*)d_in[41];

  float* ws = (float*)d_ws;
  float* xl   = ws;                              // N*HC
  float* xr   = xl  + (size_t)kN * kHC;          // N*HC
  float* hA   = xr  + (size_t)kN * kHC;          // N*HC (layer1 out)
  float* hB   = hA  + (size_t)kN * kHC;          // N*HC (layer2 out)
  float* deg  = hB  + (size_t)kN * kHC;          // N
  float* eas  = deg + kN;                        // N*8 -> ea_loop
  float* pbuf = eas + (size_t)kN * 8;            // Etot*4 (logits -> p)
  unsigned* lmax = (unsigned*)(pbuf + (size_t)kEtot * 4);  // N*4
  float* psum = (float*)(lmax + (size_t)kN * 4); // N*4
  float* xg   = psum + (size_t)kN * 4;           // G*256
  float* z    = xg   + (size_t)kG * 256;         // G*384 (concat)
  float* z1   = z    + (size_t)kG * 384;         // G*192
  float* z2   = z1   + (size_t)kG * 192;         // G*96
  _Float16* warena = (_Float16*)(z2 + (size_t)kG * 96);    // f16 weight arena

  // zero the (padded) f16 weight arena, then convert+transpose each weight
  fillu(stream, (unsigned*)warena, 0u, 1000000);  // 4 MB >= total arena
  size_t woff = 0;
  auto prep = [&](const float* W, int K, int Nout) -> WT {
    int Kp = (K + 31) & ~31, Np = (Nout + 63) & ~63;
    _Float16* dst = warena + woff;
    woff += (size_t)Kp * Np;
    k_wprep<<<(K + 31) / 32, 256, 0, stream>>>(W, dst, K, Nout, Kp);
    return WT{dst, Kp};
  };
  WT w1l = prep(c1w[0], kFin, kHC);
  WT w1r = prep(c1w[2], kFin, kHC);
  WT w2l = prep(c2w[0], kHC, kHC);
  WT w2r = prep(c2w[2], kHC, kHC);
  WT wg  = prep(gW,  256,  64);
  WT wf  = prep(fW,  2048, 256);
  WT wd  = prep(dW,  200,  64);
  WT wk1 = prep(k1W, 384,  192);
  WT wk2 = prep(k2W, 192,  96);
  WT wk3 = prep(k3W, 96,   12);

  // self-loop edge_attr mean
  fillf(stream, deg, 0.f, kN);
  fillf(stream, eas, 0.f, (long long)kN * 8);
  k_deg_ea<<<(kE + 255) / 256, 256, 0, stream>>>(ei, ea, deg, eas);
  k_ea_div<<<(kN * 8 + 255) / 256, 256, 0, stream>>>(deg, eas);

  // two GATv2 layers (+relu)
  gat_layer(stream, x,  kFin, w1l, w1r, c1w, xl, xr, hA, lmax, psum, pbuf, ei, ea, eas);
  gat_layer(stream, hA, kHC,  w2l, w2r, c2w, xl, xr, hB, lmax, psum, pbuf, ei, ea, eas);

  // global max pool
  fillf(stream, xg, 0.f, (long long)kG * 256);
  k_pool<<<((long long)kN * kHC + 255) / 256, 256, 0, stream>>>(hB, batch, xg);

  // fusion branches write straight into concat buffer z[:, off:off+w]
  gemm(stream, xg,   wg, gb, ggn, gbt, kG, 256,  64,  z,  384, 0,   2);
  gemm(stream, fp,   wf, fb, fgn, fbt, kG, 2048, 256, z,  384, 64,  2);
  gemm(stream, desc, wd, db, dgn, dbt, kG, 200,  64,  z,  384, 320, 2);
  // head MLP
  gemm(stream, z,  wk1, k1b, k1g, k1t, kG, 384, 192, z1, 192, 0, 2);
  gemm(stream, z1, wk2, k2b, k2g, k2t, kG, 192, 96,  z2, 96,  0, 2);
  gemm(stream, z2, wk3, k3b, nullptr, nullptr, kG, 96, 12, (float*)d_out, 12, 0, 0);
}
